// RNNEncoder_40089224741416
// MI455X (gfx1250) — matrix operand order repaired and checked by reading, not compiled, not run on hardware
//
#include <hip/hip_runtime.h>

// Problem constants from the reference
constexpr int Bc = 64;    // batch
constexpr int Tc = 1024;  // timesteps (sequential scan)
constexpr int Fc = 512;   // features
constexpr int K2 = 2 * Fc;        // 1024 = concat(x, h) depth
constexpr int NCOL = 2 * Fc;      // 1024 = [Wh cols | Ww cols]

typedef __attribute__((ext_vector_type(16))) __bf16 v16bf;
typedef __attribute__((ext_vector_type(8)))  __bf16 v8bf;
typedef __attribute__((ext_vector_type(8)))  float  v8f;
typedef __attribute__((ext_vector_type(4)))  float  vf4;

// ---------------------------------------------------------------------------
// Workspace layout (~2.2 MB total)
//   [0, 2MB)              : packed weights, bf16, column-major: wpack[n*1024 + k]
//   [2MB, 2MB+64KB)       : hidden state buffer A (bf16, [B, F])
//   [2MB+64KB, 2MB+128KB) : hidden state buffer B (bf16, [B, F])
//   [2MB+128KB, +8)       : global barrier {count, generation}
// ---------------------------------------------------------------------------
constexpr size_t WS_WPACK = 0;
constexpr size_t WS_H0    = (size_t)2 << 20;
constexpr size_t WS_H1    = WS_H0 + (size_t)Bc * Fc * sizeof(__bf16);
constexpr size_t WS_BAR   = WS_H1 + (size_t)Bc * Fc * sizeof(__bf16);

// LDS: 32 weight columns, padded stride to kill bank conflicts.
// 1032 bf16 = 2064 B = 516 dwords; 516 mod 64 = 4 -> lane m starts at bank
// quad (4m + c) mod 64: all 16 lanes of a B-fragment hit distinct banks.
constexpr int LDS_STRIDE = 1032;

// Pack Wh/Ww (fp32 row-major [2F, F]) into bf16 column-major so each lane's
// B-tile fragment (two runs of 8 K values of its column) is read as two 16 B loads.
__global__ void pack_weights_kernel(const float* __restrict__ Wh,
                                    const float* __restrict__ Ww,
                                    __bf16* __restrict__ wpack) {
    int idx = blockIdx.x * blockDim.x + threadIdx.x;   // 0 .. 1024*1024-1
    int n = idx >> 10;          // output column 0..1023
    int k = idx & (K2 - 1);     // depth 0..1023
    float v = (n < Fc) ? Wh[(size_t)k * Fc + n]
                       : Ww[(size_t)k * Fc + (n - Fc)];
    wpack[(size_t)n * K2 + k] = (__bf16)v;
}

// Initialize hidden state (bf16 copy of fp32 initial hidden) + reset barrier.
__global__ void init_state_kernel(const float* __restrict__ hidden,
                                  __bf16* __restrict__ h0,
                                  int* __restrict__ bar) {
    int i = blockIdx.x * blockDim.x + threadIdx.x;
    if (i < Bc * Fc) h0[i] = (__bf16)hidden[i];
    if (i == 0) { bar[0] = 0; bar[1] = 0; }
}

// Device-wide sense-reversal barrier. 32 blocks are far below WGP capacity,
// so all blocks are co-resident and forward progress is guaranteed.
__device__ __forceinline__ void global_barrier(int* cnt, int* gen, int nblocks) {
    __syncthreads();
    __threadfence();
    if (threadIdx.x == 0) {
        int g = __hip_atomic_load(gen, __ATOMIC_RELAXED, __HIP_MEMORY_SCOPE_AGENT);
        int a = __hip_atomic_fetch_add(cnt, 1, __ATOMIC_ACQ_REL, __HIP_MEMORY_SCOPE_AGENT);
        if (a == nblocks - 1) {
            __hip_atomic_store(cnt, 0, __ATOMIC_RELAXED, __HIP_MEMORY_SCOPE_AGENT);
            __hip_atomic_fetch_add(gen, 1, __ATOMIC_RELEASE, __HIP_MEMORY_SCOPE_AGENT);
        } else {
            while (__hip_atomic_load(gen, __ATOMIC_ACQUIRE, __HIP_MEMORY_SCOPE_AGENT) == g) {
                __builtin_amdgcn_s_sleep(1);
            }
        }
    }
    __syncthreads();
}

// Persistent RNN kernel: 32 blocks (one per 16-wide feature tile) x 4 waves
// (one per 16-wide batch tile). Weights for the block (32 columns, 64 KB)
// are staged into LDS ONCE and reused for all 1024 timesteps. Each wave
// accumulates BOTH GEMM products (Wh and Ww columns) for its 16x16 tile with
// v_wmma_f32_16x16x32_bf16, applies the tanh/sigmoid gating, writes fp32
// output, and publishes the done-masked next hidden state (bf16 ping-pong).
//
// A lane layout (16-bit A, 16x32, wave32):
//   lane l (l<16):  row M=l,    elems 0..7 -> K=kk+0..7,  8..15 -> K=kk+16..23
//   lane l (l>=16): row M=l-16, elems 0..7 -> K=kk+8..15, 8..15 -> K=kk+24..31
// B lane layout (16-bit B, 32x16, wave32):
//   lane l: column N=l%16, elems 0..7 -> K=kk+(l/16)*8+0..7, 8..15 -> K=kk+16+(l/16)*8+0..7 (as A)
__global__ __launch_bounds__(128, 1)
void rnn_persistent_kernel(const float* __restrict__ input,          // [B,T,F] fp32
                           const unsigned char* __restrict__ done,   // [B,T] bool(1B)
                           const float* __restrict__ bh,             // [F]
                           const float* __restrict__ bw,             // [F]
                           const __bf16* __restrict__ wpack,         // [1024,1024] col-major
                           __bf16* __restrict__ hA,                  // ping
                           __bf16* __restrict__ hB,                  // pong
                           int* __restrict__ bar,
                           float* __restrict__ out,                  // [B,T,F]
                           float* __restrict__ hfinal) {             // [B,F]
    __shared__ __bf16 ldsW[32 * LDS_STRIDE];   // 66,048 B of the 320 KB WGP LDS

    const int lane = threadIdx.x & 31;
    const int wave = threadIdx.x >> 5;       // 0..3  -> batch tile
    const int b0   = wave * 16;
    const int f0   = blockIdx.x * 16;        // 0..31 -> feature tile
    const int m    = lane & 15;              // A row / B,C column within tile
    const int hi   = lane >> 4;              // lane half selector

    // ---- One-time LDS fill: columns 0..15 = Wh[f0..f0+16), 16..31 = Ww ----
    for (int chunk = threadIdx.x; chunk < 32 * 128; chunk += blockDim.x) {
        const int col  = chunk >> 7;            // 0..31
        const int koff = (chunk & 127) * 8;     // 0..1016, 8 bf16 per chunk
        const int gcol = (col < 16) ? (f0 + col) : (Fc + f0 + (col - 16));
        v8bf v = *(const v8bf*)(wpack + (size_t)gcol * K2 + koff);
        *(v8bf*)(&ldsW[col * LDS_STRIDE + koff]) = v;
    }
    __syncthreads();

    const __bf16* ldsColH = &ldsW[(size_t)m * LDS_STRIDE];         // Wh col f0+m
    const __bf16* ldsColW = &ldsW[(size_t)(16 + m) * LDS_STRIDE];  // Ww col f0+m

    __bf16* hbuf[2] = { hA, hB };
    const int arow = b0 + m;                 // A-operand row owned by this lane
    const float bhv = bh[f0 + m];
    const float bwv = bw[f0 + m];

    for (int t = 0; t < Tc; ++t) {
        const __bf16* hread  = hbuf[t & 1];
        __bf16*       hwrite = hbuf[(t + 1) & 1];
        const float*  xrow   = input + ((size_t)arow * Tc + t) * Fc;
        const __bf16* hrow   = hread + (size_t)arow * Fc;

        v8f accH = {};
        v8f accW = {};

        // ---- K = [0, 512): x_t part (fp32 global loads, convert to bf16) ----
        #pragma unroll 4
        for (int kk = 0; kk < Fc; kk += 32) {
            const int k0 = kk + hi * 8;
            const int k1 = kk + 16 + hi * 8;
            vf4 x0 = *(const vf4*)(xrow + k0);
            vf4 x1 = *(const vf4*)(xrow + k0 + 4);
            vf4 x2 = *(const vf4*)(xrow + k1);
            vf4 x3 = *(const vf4*)(xrow + k1 + 4);
            v16bf a;
            #pragma unroll
            for (int i = 0; i < 4; ++i) {
                a[i]      = (__bf16)x0[i];
                a[4 + i]  = (__bf16)x1[i];
                a[8 + i]  = (__bf16)x2[i];
                a[12 + i] = (__bf16)x3[i];
            }
            const int bo = kk + hi * 8;
            v8bf bt0 = *(const v8bf*)(ldsColH + bo);
            v8bf bt1 = *(const v8bf*)(ldsColH + bo + 16);
            v8bf wt0 = *(const v8bf*)(ldsColW + bo);
            v8bf wt1 = *(const v8bf*)(ldsColW + bo + 16);
            v16bf bt, wt;
            #pragma unroll
            for (int i = 0; i < 8; ++i) {
                bt[i] = bt0[i]; bt[8 + i] = bt1[i];
                wt[i] = wt0[i]; wt[8 + i] = wt1[i];
            }
            accH = __builtin_amdgcn_wmma_f32_16x16x32_bf16(false, a, false, bt,
                                                           (short)0, accH, false, false);
            accW = __builtin_amdgcn_wmma_f32_16x16x32_bf16(false, a, false, wt,
                                                           (short)0, accW, false, false);
        }

        // ---- K = [512, 1024): recurrent h part (bf16 global loads) ----
        #pragma unroll 4
        for (int kk = Fc; kk < K2; kk += 32) {
            const int k0 = kk - Fc + hi * 8;
            const int k1 = kk - Fc + 16 + hi * 8;
            v8bf h0 = *(const v8bf*)(hrow + k0);
            v8bf h1 = *(const v8bf*)(hrow + k1);
            v16bf a;
            #pragma unroll
            for (int i = 0; i < 8; ++i) { a[i] = h0[i]; a[8 + i] = h1[i]; }
            const int bo = kk + hi * 8;
            v8bf bt0 = *(const v8bf*)(ldsColH + bo);
            v8bf bt1 = *(const v8bf*)(ldsColH + bo + 16);
            v8bf wt0 = *(const v8bf*)(ldsColW + bo);
            v8bf wt1 = *(const v8bf*)(ldsColW + bo + 16);
            v16bf bt, wt;
            #pragma unroll
            for (int i = 0; i < 8; ++i) {
                bt[i] = bt0[i]; bt[8 + i] = bt1[i];
                wt[i] = wt0[i]; wt[8 + i] = wt1[i];
            }
            accH = __builtin_amdgcn_wmma_f32_16x16x32_bf16(false, a, false, bt,
                                                           (short)0, accH, false, false);
            accW = __builtin_amdgcn_wmma_f32_16x16x32_bf16(false, a, false, wt,
                                                           (short)0, accW, false, false);
        }

        // ---- Epilogue: gating + output + masked hidden update ----
        // C/D layout: VGPR r of lane l holds row M = r + 8*(l/16), col N = l%16.
        const bool last = (t == Tc - 1);
        const int fcol = f0 + m;
        #pragma unroll
        for (int r = 0; r < 8; ++r) {
            const int row = b0 + hi * 8 + r;
            const size_t oi = ((size_t)row * Tc + t) * Fc + fcol;
            const float xv   = input[oi];
            const float hidv = tanhf(accH[r] + bhv);
            const float wv   = 1.0f / (1.0f + expf(-(accW[r] + bwv)));
            const float ov   = wv * xv + (1.0f - wv) * hidv;
            out[oi] = ov;
            const float hn = done[(size_t)row * Tc + t] ? 0.0f : ov;
            hwrite[(size_t)row * Fc + fcol] = (__bf16)hn;
            if (last) hfinal[(size_t)row * Fc + fcol] = hn;
        }

        // Make this step's hidden state visible to all blocks before t+1.
        global_barrier(bar, bar + 1, gridDim.x);
    }
}

extern "C" void kernel_launch(void* const* d_in, const int* in_sizes, int n_in,
                              void* d_out, int out_size, void* d_ws, size_t ws_size,
                              hipStream_t stream) {
    // setup_inputs() order: input, hidden, done, Wh, bh, Ww, bw
    const float*         input  = (const float*)d_in[0];
    const float*         hidden = (const float*)d_in[1];
    const unsigned char* done   = (const unsigned char*)d_in[2]; // jnp.bool_ -> 1 byte
    const float*         Wh     = (const float*)d_in[3];
    const float*         bh     = (const float*)d_in[4];
    const float*         Ww     = (const float*)d_in[5];
    const float*         bw     = (const float*)d_in[6];

    float* out    = (float*)d_out;                 // [B,T,F] then [B,F] h_final
    float* hfinal = out + (size_t)Bc * Tc * Fc;

    __bf16* wpack = (__bf16*)((char*)d_ws + WS_WPACK);
    __bf16* h0    = (__bf16*)((char*)d_ws + WS_H0);
    __bf16* h1    = (__bf16*)((char*)d_ws + WS_H1);
    int*    bar   = (int*)   ((char*)d_ws + WS_BAR);

    // 1) Pack weights to bf16, B-operand-friendly layout (1M elements).
    pack_weights_kernel<<<(K2 * NCOL) / 256, 256, 0, stream>>>(Wh, Ww, wpack);
    // 2) Initialize hidden state + barrier.
    init_state_kernel<<<(Bc * Fc + 255) / 256, 256, 0, stream>>>(hidden, h0, bar);
    // 3) Single persistent kernel runs the full 1024-step recurrence.
    rnn_persistent_kernel<<<32, 128, 0, stream>>>(input, done, bh, bw, wpack,
                                                  h0, h1, bar, out, hfinal);
}
